// GCNnet_43404939494160
// MI455X (gfx1250) — matrix operand order repaired and checked by reading, not compiled, not run on hardware
//
#include <hip/hip_runtime.h>

typedef __bf16 bf16;
typedef bf16 v16bf __attribute__((ext_vector_type(16)));
typedef float v8f  __attribute__((ext_vector_type(8)));
typedef unsigned int u32x4 __attribute__((ext_vector_type(4)));

#define NN 20000
#define NE 320000
#define NG 64
#define CDIV(a,b) (((a)+(b)-1)/(b))

// ---- monotone float<->uint mapping for atomic max on floats ----
__device__ __forceinline__ unsigned fmap(float f){
  unsigned u = __float_as_uint(f);
  return (u & 0x80000000u) ? ~u : (u | 0x80000000u);
}
__device__ __forceinline__ float funmap(unsigned u){
  u = (u & 0x80000000u) ? (u & 0x7FFFFFFFu) : ~u;
  return __uint_as_float(u);
}

// =====================  WMMA bf16 GEMM  =====================
// C[M,N] = A[M,K](bf16,row-major) * Bt[N,K](bf16,row-major == B^T) (+bias)(+relu)
// One 32x64 output tile per wave32 (2 M-subtiles x 4 N-subtiles of 16x16).
// Per K=32 step: 2 A-fragments + 4 B-fragments (12 b128 loads) feed 8 WMMAs.
// Requires M % 32 == 0, N % 64 == 0, K % 32 == 0 (all shapes here qualify).
__global__ void gemm_bf16_wmma(const bf16* __restrict__ A, const bf16* __restrict__ Bt,
                               float* __restrict__ C, const float* __restrict__ bias,
                               int M, int Nn, int K, int relu)
{
  int wave = blockIdx.x * (blockDim.x >> 5) + (threadIdx.x >> 5);
  int lane = threadIdx.x & 31;
  int tilesN = Nn >> 6;
  int tiles  = (M >> 5) * tilesN;
  if (wave >= tiles) return;                 // whole wave exits together (EXEC stays full)
  int tm = (wave / tilesN) << 5;
  int tn = (wave % tilesN) << 6;

  int row = lane & 15;                       // M index (A) / N index (B,C)
  int hi  = lane >> 4;                       // lane-half select
  const bf16* arow0 = A  + (size_t)(tm + row) * K + hi * 8;   // A: K {0..7|8..15},{16..23|24..31}
  const bf16* arow1 = arow0 + (size_t)16 * K;
  const bf16* brow  = Bt + (size_t)(tn + row) * K + hi * 8;   // B: K {0..7|8..15},{16..23|24..31}

  v8f acc[2][4] = {};
  for (int k0 = 0; k0 < K; k0 += 32) {
    union U { u32x4 u[2]; v16bf v; };
    U a0, a1, b[4];
    a0.u[0] = *(const u32x4*)(arow0 + k0);
    a0.u[1] = *(const u32x4*)(arow0 + k0 + 16);
    a1.u[0] = *(const u32x4*)(arow1 + k0);
    a1.u[1] = *(const u32x4*)(arow1 + k0 + 16);
#pragma unroll
    for (int j = 0; j < 4; ++j) {
      const bf16* bp = brow + (size_t)j * 16 * K + k0;
      b[j].u[0] = *(const u32x4*)(bp);
      b[j].u[1] = *(const u32x4*)(bp + 16);
    }
#pragma unroll
    for (int j = 0; j < 4; ++j) {
      acc[0][j] = __builtin_amdgcn_wmma_f32_16x16x32_bf16(false, a0.v, false, b[j].v,
                                                          (short)0, acc[0][j], false, false);
      acc[1][j] = __builtin_amdgcn_wmma_f32_16x16x32_bf16(false, a1.v, false, b[j].v,
                                                          (short)0, acc[1][j], false, false);
    }
  }
#pragma unroll
  for (int j = 0; j < 4; ++j) {
    int n = tn + j * 16 + row;
    float bv = bias ? bias[n] : 0.0f;
#pragma unroll
    for (int mi = 0; mi < 2; ++mi) {
      float* cp = C + (size_t)(tm + mi * 16 + hi * 8) * Nn + n;
#pragma unroll
      for (int r = 0; r < 8; ++r) {
        float v = acc[mi][j][r] + bv;
        if (relu) v = fmaxf(v, 0.0f);
        cp[(size_t)r * Nn] = v;
      }
    }
  }
}

// =====================  elementwise / prep  =====================
__global__ void k_cvt_bf16(const float* __restrict__ in, bf16* __restrict__ out, int n){
  int i = blockIdx.x * blockDim.x + threadIdx.x;
  if (i < n) out[i] = (bf16)in[i];
}
// W[K,N] f32 -> Wt[N,K] bf16
__global__ void k_wt(const float* __restrict__ W, bf16* __restrict__ Wt, int K, int Nn){
  int i = blockIdx.x * blockDim.x + threadIdx.x;
  if (i >= K * Nn) return;
  int k = i / Nn, n = i % Nn;
  Wt[(size_t)n * K + k] = (bf16)W[i];
}
__global__ void k_fill_f(float* p, float v, int n){
  int i = blockIdx.x * blockDim.x + threadIdx.x; if (i < n) p[i] = v;
}
__global__ void k_fill_u(unsigned* p, unsigned v, int n){
  int i = blockIdx.x * blockDim.x + threadIdx.x; if (i < n) p[i] = v;
}

// =====================  GAT  =====================
// a_s[n,h] = <hpre[n,h,:], att_src[h,:]>, same for a_d
__global__ void k_att(const float* __restrict__ hpre, const float* __restrict__ att_src,
                      const float* __restrict__ att_dst, float* __restrict__ as_,
                      float* __restrict__ ad_, int n){
  int i = blockIdx.x * blockDim.x + threadIdx.x;
  if (i >= n * 8) return;
  int node = i >> 3, h = i & 7;
  const float* hp = hpre + (size_t)node * 256 + h * 32;
  float s0 = 0.f, s1 = 0.f;
#pragma unroll 8
  for (int c = 0; c < 32; ++c) { s0 += hp[c] * att_src[h*32+c]; s1 += hp[c] * att_dst[h*32+c]; }
  as_[i] = s0; ad_[i] = s1;
}
// init max with the self-loop edge value
__global__ void k_minit(const float* __restrict__ as_, const float* __restrict__ ad_,
                        unsigned* __restrict__ m_u, float* __restrict__ selfe, int n8){
  int i = blockIdx.x * blockDim.x + threadIdx.x;
  if (i >= n8) return;
  float e = as_[i] + ad_[i];
  e = e > 0.f ? e : 0.2f * e;
  selfe[i] = e;
  m_u[i] = fmap(e);
}
__global__ void k_epass1(const int* __restrict__ src, const int* __restrict__ dst,
                         const float* __restrict__ as_, const float* __restrict__ ad_,
                         float* __restrict__ ebuf, unsigned* __restrict__ m_u, int ne){
  int i = blockIdx.x * blockDim.x + threadIdx.x;
  if (i >= ne * 8) return;
  int e = i >> 3, h = i & 7;
  int s = src[e], d = dst[e];
  float v = as_[s*8+h] + ad_[d*8+h];
  v = v > 0.f ? v : 0.2f * v;
  ebuf[i] = v;
  atomicMax(&m_u[d*8+h], fmap(v));
}
// unmap max in place; seed sum with self-loop exp
__global__ void k_sinit(unsigned* __restrict__ m_u, const float* __restrict__ selfe,
                        float* __restrict__ ssum, float* __restrict__ selfex, int n8){
  int i = blockIdx.x * blockDim.x + threadIdx.x;
  if (i >= n8) return;
  float mm = funmap(m_u[i]);
  ((float*)m_u)[i] = mm;
  float ex = expf(selfe[i] - mm);
  selfex[i] = ex; ssum[i] = ex;
}
__global__ void k_epass2(const int* __restrict__ dst, float* __restrict__ ebuf,
                         const float* __restrict__ m_f, float* __restrict__ ssum, int ne){
  int i = blockIdx.x * blockDim.x + threadIdx.x;
  if (i >= ne * 8) return;
  int e = i >> 3, h = i & 7;
  int d = dst[e];
  float ex = expf(ebuf[i] - m_f[d*8+h]);
  ebuf[i] = ex;
  atomicAdd(&ssum[d*8+h], ex);
}
// out[n,h,c] = hpre[n,h,c]*alpha_self + b_gat  (self-loop contribution + bias)
__global__ void k_gatinit(const float* __restrict__ hpre, const float* __restrict__ selfex,
                          const float* __restrict__ ssum, const float* __restrict__ bgat,
                          float* __restrict__ out, int n){
  int node = blockIdx.x; if (node >= n) return;
  int t = threadIdx.x, h = t >> 5;
  float alpha = selfex[node*8+h] / ssum[node*8+h];
  out[(size_t)node*256 + t] = hpre[(size_t)node*256 + t] * alpha + bgat[t];
}
__global__ void k_epass3(const int* __restrict__ src, const int* __restrict__ dst,
                         const float* __restrict__ hpre, const float* __restrict__ ebuf,
                         const float* __restrict__ ssum, float* __restrict__ out, int ne){
  int e = blockIdx.x; if (e >= ne) return;
  int t = threadIdx.x, h = t >> 5;
  int s = src[e], d = dst[e];
  float alpha = ebuf[e*8+h] / ssum[d*8+h];
  atomicAdd(&out[(size_t)d*256 + t], hpre[(size_t)s*256 + t] * alpha);
}
__global__ void k_relu_cvt(const float* __restrict__ in, bf16* __restrict__ outb, int n){
  int i = blockIdx.x * blockDim.x + threadIdx.x;
  if (i < n) outb[i] = (bf16)fmaxf(in[i], 0.f);
}

// =====================  GCN  =====================
__global__ void k_deg(const int* __restrict__ dst, const float* __restrict__ ew,
                      float* __restrict__ deg, int ne){
  int i = blockIdx.x * blockDim.x + threadIdx.x;
  if (i < ne) atomicAdd(&deg[dst[i]], ew[i]);
}
__global__ void k_dinv(float* deg, int n){
  int i = blockIdx.x * blockDim.x + threadIdx.x;
  if (i < n) deg[i] = rsqrtf(deg[i]);           // deg >= 1 (self loop)
}
__global__ void k_norm(const int* __restrict__ src, const int* __restrict__ dst,
                       const float* __restrict__ ew, const float* __restrict__ dinv,
                       float* __restrict__ nrm, int ne){
  int i = blockIdx.x * blockDim.x + threadIdx.x;
  if (i < ne) nrm[i] = dinv[src[i]] * ew[i] * dinv[dst[i]];
}
// a = bias + t * dinv^2  (self-loop term)
__global__ void k_agginit(const float* __restrict__ t, const float* __restrict__ dinv,
                          const float* __restrict__ bias, float* __restrict__ a, int n, int F){
  int i = blockIdx.x * blockDim.x + threadIdx.x;
  if (i >= n * F) return;
  int node = i / F, f = i - node * F;
  float di = dinv[node];
  a[i] = bias[f] + t[i] * di * di;
}
__global__ void k_eagg(const int* __restrict__ src, const int* __restrict__ dst,
                       const float* __restrict__ t, const float* __restrict__ nrm,
                       float* __restrict__ a, int ne, int F){
  int e = blockIdx.x; if (e >= ne) return;
  int s = src[e], d = dst[e];
  float w = nrm[e];
  const float* ts = t + (size_t)s * F;
  float* ad = a + (size_t)d * F;
  for (int f = threadIdx.x; f < F; f += blockDim.x)
    atomicAdd(&ad[f], ts[f] * w);
}
__global__ void k_bnrelu_cvt(const float* __restrict__ a, const float* __restrict__ g,
                             const float* __restrict__ be, const float* __restrict__ mu,
                             const float* __restrict__ var, bf16* __restrict__ outb,
                             int n, int F){
  int i = blockIdx.x * blockDim.x + threadIdx.x;
  if (i >= n * F) return;
  int f = i % F;
  float v = g[f] * (a[i] - mu[f]) * rsqrtf(var[f] + 1e-5f) + be[f];
  outb[i] = (bf16)fmaxf(v, 0.f);
}
__global__ void k_bnrelu_res(float* __restrict__ a, const float* __restrict__ g,
                             const float* __restrict__ be, const float* __restrict__ mu,
                             const float* __restrict__ var, const float* __restrict__ res,
                             int n, int F){
  int i = blockIdx.x * blockDim.x + threadIdx.x;
  if (i >= n * F) return;
  int f = i % F;
  float v = g[f] * (a[i] - mu[f]) * rsqrtf(var[f] + 1e-5f) + be[f];
  a[i] = fmaxf(v, 0.f) + res[i];
}

// =====================  pooling + head  =====================
__global__ void k_pool(const float* __restrict__ hfin, const int* __restrict__ batch,
                       unsigned* __restrict__ pu, int n){
  int node = blockIdx.x; if (node >= n) return;
  int b = batch[node];
  for (int f = threadIdx.x; f < 1024; f += blockDim.x)
    atomicMax(&pu[b*1024 + f], fmap(hfin[(size_t)node*1024 + f]));
}
__global__ void k_unmap_cvt(const unsigned* __restrict__ pu, bf16* __restrict__ pb, int n){
  int i = blockIdx.x * blockDim.x + threadIdx.x;
  if (i < n) pb[i] = (bf16)funmap(pu[i]);
}
__global__ void k_head2(const float* __restrict__ z, const float* __restrict__ Wf2,
                        const float* __restrict__ bf2, float* __restrict__ out){
  int i = threadIdx.x; if (i >= NG * 2) return;
  int g = i >> 1, j = i & 1;
  float s = bf2[j];
  for (int k = 0; k < 256; ++k) s += z[g*256 + k] * Wf2[k*2 + j];
  out[i] = s;
}

// =====================  launch  =====================
extern "C" void kernel_launch(void* const* d_in, const int* in_sizes, int n_in,
                              void* d_out, int out_size, void* d_ws, size_t ws_size,
                              hipStream_t stream) {
  const float* x       = (const float*)d_in[0];
  const int*   ei      = (const int*)  d_in[1];
  const int*   src     = ei;
  const int*   dst     = ei + NE;
  const float* ew      = (const float*)d_in[2];
  const int*   batch   = (const int*)  d_in[3];
  const float* W_gat   = (const float*)d_in[4];
  const float* att_src = (const float*)d_in[5];
  const float* att_dst = (const float*)d_in[6];
  const float* b_gat   = (const float*)d_in[7];
  const float* W_res   = (const float*)d_in[8];
  const float* b_res   = (const float*)d_in[9];
  const float* W2      = (const float*)d_in[10];
  const float* b2      = (const float*)d_in[11];
  const float* g1      = (const float*)d_in[12];
  const float* be1     = (const float*)d_in[13];
  const float* m1      = (const float*)d_in[14];
  const float* v1      = (const float*)d_in[15];
  const float* W3      = (const float*)d_in[16];
  const float* b3      = (const float*)d_in[17];
  const float* g2      = (const float*)d_in[18];
  const float* be2     = (const float*)d_in[19];
  const float* m2      = (const float*)d_in[20];
  const float* v2      = (const float*)d_in[21];
  const float* Wf1     = (const float*)d_in[22];
  const float* bf1     = (const float*)d_in[23];
  const float* Wf2     = (const float*)d_in[24];
  const float* bf2     = (const float*)d_in[25];

  // ---- workspace carve-up ----
  char* ws = (char*)d_ws;
  size_t off = 0;
  auto take = [&](size_t bytes) -> char* {
    char* p = ws + off;
    off += (bytes + 255) & ~(size_t)255;
    return p;
  };
  // persistent
  bf16*  xb     = (bf16*)take((size_t)NN * 32 * 2);
  bf16*  Wgt_t  = (bf16*)take(256 * 32 * 2);
  bf16*  Wres_t = (bf16*)take((size_t)1024 * 256 * 2);
  bf16*  W2t    = (bf16*)take((size_t)512 * 256 * 2);
  bf16*  W3t    = (bf16*)take((size_t)1024 * 512 * 2);
  bf16*  Wf1t   = (bf16*)take((size_t)256 * 1024 * 2);
  bf16*  hb     = (bf16*)take((size_t)NN * 256 * 2);
  float* res    = (float*)take((size_t)NN * 1024 * 4);
  float* dinv   = (float*)take((size_t)NN * 4);
  float* nrm    = (float*)take((size_t)NE * 4);
  bf16*  h2b    = (bf16*)take((size_t)NN * 512 * 2);
  unsigned* pu  = (unsigned*)take((size_t)NG * 1024 * 4);
  bf16*  pooledb= (bf16*)take((size_t)NG * 1024 * 2);
  float* zbuf   = (float*)take((size_t)NG * 256 * 4);
  // region A: GAT scratch, then GCN gemm output tbuf
  char* regA = take((size_t)NN * 1024 * 4);
  float*    hpre   = (float*)regA;                                  // NN*256
  float*    ebuf   = (float*)(regA + (size_t)NN*256*4);             // NE*8
  float*    as_    = (float*)(regA + (size_t)NN*256*4 + (size_t)NE*8*4);
  float*    ad_    = as_ + (size_t)NN*8;
  unsigned* m_s    = (unsigned*)(ad_ + (size_t)NN*8);               // max, unmapped in place
  float*    ssum   = (float*)(m_s + (size_t)NN*8);
  float*    selfe  = ssum + (size_t)NN*8;
  float*    selfex = selfe + (size_t)NN*8;
  float*    tbuf   = (float*)regA;                                  // reused after GAT
  // region B: GAT output, then GCN aggregation / final features
  char* regB = take((size_t)NN * 1024 * 4);
  float* gato = (float*)regB;    // NN*256
  float* abuf = (float*)regB;    // NN*512 / NN*1024

  const int T = 256;
  auto gemm = [&](const bf16* A, const bf16* Bt, float* C, const float* bias,
                  int M, int Nn, int K, int relu) {
    int tiles = (M / 32) * (Nn / 64);
    gemm_bf16_wmma<<<CDIV(tiles, 8), 256, 0, stream>>>(A, Bt, C, bias, M, Nn, K, relu);
  };

  // ---- prep: bf16 conversions / weight transposes ----
  k_cvt_bf16<<<CDIV(NN*32, T), T, 0, stream>>>(x, xb, NN*32);
  k_wt<<<CDIV(32*256, T),   T, 0, stream>>>(W_gat, Wgt_t, 32, 256);
  k_wt<<<CDIV(256*1024, T), T, 0, stream>>>(W_res, Wres_t, 256, 1024);
  k_wt<<<CDIV(256*512, T),  T, 0, stream>>>(W2, W2t, 256, 512);
  k_wt<<<CDIV(512*1024, T), T, 0, stream>>>(W3, W3t, 512, 1024);
  k_wt<<<CDIV(1024*256, T), T, 0, stream>>>(Wf1, Wf1t, 1024, 256);

  // ---- GAT ----
  gemm(xb, Wgt_t, hpre, nullptr, NN, 256, 32, 0);
  k_att<<<CDIV(NN*8, T), T, 0, stream>>>(hpre, att_src, att_dst, as_, ad_, NN);
  k_minit<<<CDIV(NN*8, T), T, 0, stream>>>(as_, ad_, m_s, selfe, NN*8);
  k_epass1<<<CDIV(NE*8, T), T, 0, stream>>>(src, dst, as_, ad_, ebuf, m_s, NE);
  k_sinit<<<CDIV(NN*8, T), T, 0, stream>>>(m_s, selfe, ssum, selfex, NN*8);
  k_epass2<<<CDIV(NE*8, T), T, 0, stream>>>(dst, ebuf, (const float*)m_s, ssum, NE);
  k_gatinit<<<NN, 256, 0, stream>>>(hpre, selfex, ssum, b_gat, gato, NN);
  k_epass3<<<NE, 256, 0, stream>>>(src, dst, hpre, ebuf, ssum, gato, NE);
  k_relu_cvt<<<CDIV(NN*256, T), T, 0, stream>>>(gato, hb, NN*256);

  // ---- residual Linear(256 -> 1024) ----
  gemm(hb, Wres_t, res, b_res, NN, 1024, 256, 0);

  // ---- symmetric GCN normalization ----
  k_fill_f<<<CDIV(NN, T), T, 0, stream>>>(dinv, 1.0f, NN);      // self-loop weight
  k_deg<<<CDIV(NE, T), T, 0, stream>>>(dst, ew, dinv, NE);
  k_dinv<<<CDIV(NN, T), T, 0, stream>>>(dinv, NN);
  k_norm<<<CDIV(NE, T), T, 0, stream>>>(src, dst, ew, dinv, nrm, NE);

  // ---- GCN1 (256 -> 512) + BN + ReLU ----
  gemm(hb, W2t, tbuf, nullptr, NN, 512, 256, 0);
  k_agginit<<<CDIV(NN*512, T), T, 0, stream>>>(tbuf, dinv, b2, abuf, NN, 512);
  k_eagg<<<NE, 256, 0, stream>>>(src, dst, tbuf, nrm, abuf, NE, 512);
  k_bnrelu_cvt<<<CDIV(NN*512, T), T, 0, stream>>>(abuf, g1, be1, m1, v1, h2b, NN, 512);

  // ---- GCN2 (512 -> 1024) + BN + ReLU + residual ----
  gemm(h2b, W3t, tbuf, nullptr, NN, 1024, 512, 0);
  k_agginit<<<CDIV(NN*1024, T), T, 0, stream>>>(tbuf, dinv, b3, abuf, NN, 1024);
  k_eagg<<<NE, 256, 0, stream>>>(src, dst, tbuf, nrm, abuf, NE, 1024);
  k_bnrelu_res<<<CDIV(NN*1024, T), T, 0, stream>>>(abuf, g2, be2, m2, v2, res, NN, 1024);

  // ---- global max pool over graphs ----
  k_fill_u<<<CDIV(NG*1024, T), T, 0, stream>>>(pu, 0x00800000u /* fmap(-FLT_MAX) */, NG*1024);
  k_pool<<<NN, 256, 0, stream>>>(abuf, batch, pu, NN);
  k_unmap_cvt<<<CDIV(NG*1024, T), T, 0, stream>>>(pu, pooledb, NG*1024);

  // ---- head ----
  gemm(pooledb, Wf1t, zbuf, bf1, NG, 256, 1024, 1);
  k_head2<<<1, 128, 0, stream>>>(zbuf, Wf2, bf2, (float*)d_out);
}
